// RSSM_71579924955486
// MI455X (gfx1250) — matrix operand order repaired and checked by reading, not compiled, not run on hardware
//
#include <hip/hip_runtime.h>
#include <hip/hip_bf16.h>
#include <cstddef>

#define B_ 256
#define T_ 64
#define STOCH_ 128
#define DET_ 512
#define HID_ 512
#define OBS_ 1024
#define ACT_ 32
#define OUTC_ 1152

typedef __attribute__((ext_vector_type(16))) __bf16 v16bf;
typedef __attribute__((ext_vector_type(8)))  float  v8f;
typedef __attribute__((ext_vector_type(4)))  unsigned int v4u;

union BV32 { v4u u[2]; v16bf bf; };

__device__ __forceinline__ unsigned short f2bf(float f) {
  unsigned int u = __float_as_uint(f);
  u += 0x7FFFu + ((u >> 16) & 1u);           // round-to-nearest-even
  return (unsigned short)(u >> 16);
}
__device__ __forceinline__ float eluf(float v)      { return v > 0.f ? v : __expf(v) - 1.f; }
__device__ __forceinline__ float sigf(float v)      { return 1.f / (1.f + __expf(-v)); }
__device__ __forceinline__ float softplusf(float v) { return fmaxf(v, 0.f) + log1pf(__expf(-fabsf(v))); }

// ---- WMMA operand loaders -------------------------------------------------
// A (16x32 bf16, row-major in LDS, contiguous K order matching B tiles):
//   lanes 0-15 : M=lane, K = 0..15;  lanes 16-31: M=lane-16, K = 16..31
__device__ __forceinline__ v16bf load_a(const unsigned short* __restrict__ a,
                                        int stride, int kt, int lane) {
  const int row = lane & 15;
  const int kb  = (lane >> 4) << 4;          // 0 or 16
  const unsigned short* p = a + row * stride + kt * 32 + kb;
  BV32 v;
  v.u[0] = *(const v4u*)(p);                 // K = kb..kb+7
  v.u[1] = *(const v4u*)(p + 8);             // K = kb+8..kb+15
  return v.bf;
}
// B (32x16 bf16): pre-swizzled lane-major tiles; each lane reads its 16
// contiguous bf16 (2 x b128). Contiguous K order (matches load_a): lanes 0-15 K=0..15, lanes 16-31 K=16..31.
__device__ __forceinline__ v16bf load_b(const unsigned short* __restrict__ w,
                                        int ntiles, int kt, int nt, int lane) {
  const unsigned short* p = w + (((size_t)kt * ntiles + nt) << 9) + (lane << 4);
  BV32 v;
  v.u[0] = *(const v4u*)(p);
  v.u[1] = *(const v4u*)(p + 8);
  return v.bf;
}
__device__ __forceinline__ v8f mm_acc(v8f acc, const unsigned short* __restrict__ aLds,
                                      int stride, const unsigned short* __restrict__ w,
                                      int ntiles, int nt, int ktiles, int lane) {
#pragma unroll 4
  for (int kt = 0; kt < ktiles; ++kt) {
    v16bf a = load_a(aLds, stride, kt, lane);
    v16bf b = load_b(w, ntiles, kt, nt, lane);
    acc = __builtin_amdgcn_wmma_f32_16x16x32_bf16(false, a, false, b, (short)0, acc, false, false);
  }
  return acc;
}

// ---- Weight swizzle: fp32 row-major [K,N] -> bf16 B-tiles (32x16, lane-major)
__global__ void swizzle_w(const float* __restrict__ W, unsigned short* __restrict__ dst,
                          int K, int N) {
  int idx = blockIdx.x * blockDim.x + threadIdx.x;
  if (idx >= K * N) return;
  int tile = idx >> 9;
  int rem  = idx & 511;
  int lane = rem >> 4;
  int i    = rem & 15;
  int ntiles = N >> 4;
  int kt = tile / ntiles;
  int nt = tile - kt * ntiles;
  int k  = kt * 32 + ((lane >> 4) << 4) + i;
  int n  = nt * 16 + (lane & 15);
  dst[idx] = f2bf(W[(size_t)k * N + n]);
}

// ---- act_pre[bt,n] = actions[bt,:] @ prior_w1[0:32,:] + prior_b1  (K=32, trivial)
__global__ void act_pre_kernel(const float* __restrict__ actions,
                               const float* __restrict__ w1,
                               const float* __restrict__ b1,
                               float* __restrict__ actp) {
  int idx = blockIdx.x * blockDim.x + threadIdx.x;
  if (idx >= B_ * T_ * 512) return;
  int n  = idx & 511;
  int bt = idx >> 9;
  float acc = b1[n];
  const float* a = actions + (size_t)bt * ACT_;
#pragma unroll
  for (int k = 0; k < ACT_; ++k) acc += a[k] * w1[k * 512 + n];
  actp[idx] = acc;
}

// ---- obs_pre[bt,n] = obs[bt,:] @ post_w1[512:1536,:] + post_b1  (WMMA, K=1024)
__global__ void __launch_bounds__(256) obs_pre_kernel(const float* __restrict__ obs,
                                                      const unsigned short* __restrict__ w1o,
                                                      const float* __restrict__ b1,
                                                      float* __restrict__ obsp) {
  extern __shared__ unsigned short sA[];     // 16 x 1024 bf16 (32 KB)
  const int mt   = blockIdx.x;
  const int tid  = threadIdx.x;
  const int wave = tid >> 5;
  const int lane = tid & 31;
  for (int i = tid; i < 16 * OBS_; i += 256) {
    int m = i >> 10, k = i & 1023;
    sA[i] = f2bf(obs[((size_t)(mt * 16 + m)) * OBS_ + k]);
  }
  __syncthreads();
  const int col = lane & 15, mb = (lane >> 4) << 3;
  for (int nt = wave; nt < 32; nt += 8) {
    v8f acc = {};
    acc = mm_acc(acc, sA, OBS_, w1o, 32, nt, 32, lane);
    int n = nt * 16 + col;
#pragma unroll
    for (int r = 0; r < 8; ++r)
      obsp[((size_t)(mt * 16 + r + mb)) * 512 + n] = acc[r] + b1[n];
  }
}

// ---- Persistent scan kernel: 16 WGs x 16 waves, 16 batch rows each ---------
__global__ void __launch_bounds__(512, 1) rssm_scan(
    const float* __restrict__ act_pre, const float* __restrict__ obs_pre,
    const float* __restrict__ noise_post,
    const unsigned short* __restrict__ w1s, const unsigned short* __restrict__ wi,
    const unsigned short* __restrict__ wh,  const float* __restrict__ gru_b,
    const unsigned short* __restrict__ w2,  const float* __restrict__ b2,
    const unsigned short* __restrict__ w3,  const float* __restrict__ b3,
    const unsigned short* __restrict__ w1d, const unsigned short* __restrict__ w2p,
    const float* __restrict__ b2p, float* __restrict__ out) {
  extern __shared__ char smem[];
  float* s_h             = (float*)(smem);            // 16x512 f32  det state
  float* s_ax            = (float*)(smem + 32768);    // 16x512 f32  a_x cache / qm,qs staging
  float* s_z             = (float*)(smem + 65536);    // 16x512 f32  update gate
  unsigned short* s_st   = (unsigned short*)(smem + 98304);   // 16x128 bf16 stoch
  unsigned short* s_hbf  = (unsigned short*)(smem + 102400);  // 16x512 bf16 det
  unsigned short* s_x    = (unsigned short*)(smem + 118784);  // 16x512 bf16 activation
  unsigned short* s_rh   = (unsigned short*)(smem + 135168);  // 16x512 bf16 r*h / x3

  const int tid   = threadIdx.x;
  const int wave  = tid >> 5;
  const int lane  = tid & 31;
  const int col   = lane & 15;
  const int mb    = (lane >> 4) << 3;
  const int bbase = blockIdx.x * 16;

  for (int i = tid; i < 16 * DET_; i += 512) { s_h[i] = 0.f; s_hbf[i] = 0; }
  for (int i = tid; i < 16 * STOCH_; i += 512) s_st[i] = 0;
  __syncthreads();

  for (int t = 0; t < T_; ++t) {
    // prefetch next step's streamed activations (global_prefetch_b8)
    if (t + 1 < T_) {
      int r = tid & 255, m = r >> 4, ln = r & 15;
      const float* base = (tid < 256) ? act_pre : obs_pre;
      const char* p = (const char*)(base + (((size_t)(bbase + m) * T_ + t + 1) << 9)) + ln * 128;
      __builtin_prefetch(p, 0, 1);
      if (tid < 64) {
        const char* q = (const char*)(noise_post + ((size_t)(bbase + (tid >> 2)) * T_ + t + 1) * STOCH_)
                        + (tid & 3) * 128;
        __builtin_prefetch(q, 0, 1);
      }
    }

    // Phase 1: x = elu(act_pre + stoch @ W1s)           N=512 K=128
    for (int nt = wave; nt < 32; nt += 16) {
      v8f acc = {};
      acc = mm_acc(acc, s_st, STOCH_, w1s, 32, nt, 4, lane);
      int n = nt * 16 + col;
#pragma unroll
      for (int r = 0; r < 8; ++r) {
        int m = r + mb;
        float v = acc[r] + act_pre[(((size_t)(bbase + m) * T_ + t) << 9) + n];
        s_x[m * 512 + n] = f2bf(eluf(v));
      }
    }
    __syncthreads();

    // Phase 2: gx = x@Wi (N=1536); zr += h@Wh[:, :1024]; split z / r*h / a_x
    for (int nt = wave; nt < 96; nt += 16) {
      v8f acc = {};
      acc = mm_acc(acc, s_x, DET_, wi, 96, nt, 16, lane);
      if (nt < 64) acc = mm_acc(acc, s_hbf, DET_, wh, 96, nt, 16, lane);
      int n = nt * 16 + col;
#pragma unroll
      for (int r = 0; r < 8; ++r) {
        int m = r + mb;
        float v = acc[r] + gru_b[n];
        if (n < 512) {
          s_z[m * 512 + n] = sigf(v);
        } else if (n < 1024) {
          int c = n - 512;
          s_rh[m * 512 + c] = f2bf(sigf(v) * s_h[m * 512 + c]);
        } else {
          s_ax[m * 512 + (n - 1024)] = v;
        }
      }
    }
    __syncthreads();

    // Phase 3: a = tanh(a_x + (r*h)@Wh[:,1024:]); h = (1-z)h + z a   N=512 K=512
    for (int nt = wave; nt < 32; nt += 16) {
      v8f acc = {};
      acc = mm_acc(acc, s_rh, DET_, wh, 96, nt + 64, 16, lane);
      int n = nt * 16 + col;
#pragma unroll
      for (int r = 0; r < 8; ++r) {
        int m = r + mb;
        float a  = tanhf(s_ax[m * 512 + n] + acc[r]);
        float z  = s_z[m * 512 + n];
        float hn = (1.f - z) * s_h[m * 512 + n] + z * a;
        s_h[m * 512 + n]   = hn;
        s_hbf[m * 512 + n] = f2bf(hn);
        out[((size_t)(bbase + m) * T_ + t) * OUTC_ + 128 + n] = hn;
      }
    }
    __syncthreads();

    // Phase 4: x2 = elu(h @ W2 + b2)                    N=512 K=512
    for (int nt = wave; nt < 32; nt += 16) {
      v8f acc = {};
      acc = mm_acc(acc, s_hbf, DET_, w2, 32, nt, 16, lane);
      int n = nt * 16 + col;
#pragma unroll
      for (int r = 0; r < 8; ++r)
        s_x[(r + mb) * 512 + n] = f2bf(eluf(acc[r] + b2[n]));
    }
    __syncthreads();

    // Phase 5: prior stats = x2 @ W3 + b3 (N=256); Phase 6: x3 = elu(h@W1d + obs_pre)
    for (int nt = wave; nt < 16; nt += 16) {
      v8f acc = {};
      acc = mm_acc(acc, s_x, DET_, w3, 16, nt, 16, lane);
      int n = nt * 16 + col;
#pragma unroll
      for (int r = 0; r < 8; ++r) {
        int m = r + mb;
        size_t ob = ((size_t)(bbase + m) * T_ + t) * OUTC_;
        float v = acc[r] + b3[n];
        if (n < 128) out[ob + 640 + n] = v;
        else         out[ob + 768 + (n - 128)] = softplusf(v) + 0.1f;
      }
    }
    for (int nt = wave; nt < 32; nt += 16) {
      v8f acc = {};
      acc = mm_acc(acc, s_hbf, DET_, w1d, 32, nt, 16, lane);
      int n = nt * 16 + col;
#pragma unroll
      for (int r = 0; r < 8; ++r) {
        int m = r + mb;
        float v = acc[r] + obs_pre[(((size_t)(bbase + m) * T_ + t) << 9) + n];
        s_rh[m * 512 + n] = f2bf(eluf(v));
      }
    }
    __syncthreads();

    // Phase 7: post stats = x3 @ W2p + b2p (N=256); stage qm/qs
    for (int nt = wave; nt < 16; nt += 16) {
      v8f acc = {};
      acc = mm_acc(acc, s_rh, DET_, w2p, 16, nt, 16, lane);
      int n = nt * 16 + col;
#pragma unroll
      for (int r = 0; r < 8; ++r) {
        int m = r + mb;
        size_t ob = ((size_t)(bbase + m) * T_ + t) * OUTC_;
        float v = acc[r] + b2p[n];
        if (n < 128) { out[ob + 896 + n] = v; s_ax[m * 256 + n] = v; }
        else { float qs = softplusf(v) + 0.1f; out[ob + 1024 + (n - 128)] = qs; s_ax[m * 256 + n] = qs; }
      }
    }
    __syncthreads();

    // Phase 8: stoch = qm + qs * noise_post ; write feat[0:128]
    for (int i = tid; i < 16 * STOCH_; i += 512) {
      int m = i >> 7, k = i & 127;
      size_t bt = (size_t)(bbase + m) * T_ + t;
      float st = s_ax[m * 256 + k] + s_ax[m * 256 + 128 + k] * noise_post[bt * STOCH_ + k];
      out[bt * OUTC_ + k] = st;
      s_st[i] = f2bf(st);
    }
    __syncthreads();
  }
}

extern "C" void kernel_launch(void* const* d_in, const int* in_sizes, int n_in,
                              void* d_out, int out_size, void* d_ws, size_t ws_size,
                              hipStream_t stream) {
  (void)in_sizes; (void)n_in; (void)out_size; (void)ws_size;
  const float* observations = (const float*)d_in[0];
  const float* actions      = (const float*)d_in[1];
  const float* noise_post   = (const float*)d_in[3];   // noise_prior (d_in[2]) is dead in reference
  const float* prior_w1     = (const float*)d_in[4];
  const float* prior_b1     = (const float*)d_in[5];
  const float* gru_wi       = (const float*)d_in[6];
  const float* gru_wh       = (const float*)d_in[7];
  const float* gru_b        = (const float*)d_in[8];
  const float* prior_w2     = (const float*)d_in[9];
  const float* prior_b2     = (const float*)d_in[10];
  const float* prior_w3     = (const float*)d_in[11];
  const float* prior_b3     = (const float*)d_in[12];
  const float* post_w1      = (const float*)d_in[13];
  const float* post_b1      = (const float*)d_in[14];
  const float* post_w2      = (const float*)d_in[15];
  const float* post_b2      = (const float*)d_in[16];
  float* out = (float*)d_out;

  char* ws = (char*)d_ws;
  float* act_pre = (float*)(ws);                        // 33,554,432 B
  float* obs_pre = (float*)(ws + 33554432);             // 33,554,432 B
  unsigned short* w1s = (unsigned short*)(ws + 67108864);  // 128x512
  unsigned short* wi  = (unsigned short*)(ws + 67239936);  // 512x1536
  unsigned short* wh  = (unsigned short*)(ws + 68812800);  // 512x1536
  unsigned short* w2  = (unsigned short*)(ws + 70385664);  // 512x512
  unsigned short* w3  = (unsigned short*)(ws + 70909952);  // 512x256
  unsigned short* w1d = (unsigned short*)(ws + 71172096);  // 512x512
  unsigned short* w1o = (unsigned short*)(ws + 71696384);  // 1024x512
  unsigned short* w2p = (unsigned short*)(ws + 72744960);  // 512x256

  auto swz = [&](const float* src, unsigned short* dst, int K, int N) {
    int total = K * N;
    hipLaunchKernelGGL(swizzle_w, dim3((total + 255) / 256), dim3(256), 0, stream, src, dst, K, N);
  };
  swz(prior_w1 + 32 * 512, w1s, 128, 512);   // stoch part of prior_w1
  swz(gru_wi, wi, 512, 1536);
  swz(gru_wh, wh, 512, 1536);
  swz(prior_w2, w2, 512, 512);
  swz(prior_w3, w3, 512, 256);
  swz(post_w1, w1d, 512, 512);               // det part
  swz(post_w1 + 512 * 512, w1o, 1024, 512);  // obs part
  swz(post_w2, w2p, 512, 256);

  hipLaunchKernelGGL(act_pre_kernel, dim3(B_ * T_ * 512 / 256), dim3(256), 0, stream,
                     actions, prior_w1, prior_b1, act_pre);
  hipLaunchKernelGGL(obs_pre_kernel, dim3(B_ * T_ / 16), dim3(256), 32768, stream,
                     observations, w1o, post_b1, obs_pre);
  hipLaunchKernelGGL(rssm_scan, dim3(16), dim3(512), 151552, stream,
                     act_pre, obs_pre, noise_post, w1s, wi, wh, gru_b,
                     w2, prior_b2, w3, prior_b3, w1d, w2p, post_b2, out);
}
